// Block_85598698209846
// MI455X (gfx1250) — matrix operand order repaired and checked by reading, not compiled, not run on hardware
//
#include <hip/hip_runtime.h>
#include <hip/hip_bf16.h>
#include <math.h>

// ---------------- problem constants ----------------
#define Bq   2
#define Sq   2048
#define Dq   1024
#define NHq  16
#define NKVq 4
#define HDq  64
#define KVDq (NKVq*HDq)      // 256
#define NEq  8
#define INTERq 512           // MM*D/NE = 4*1024/8
#define Mq   (Bq*Sq)         // 4096 tokens
#define EPSq 1.1920929e-07f

typedef __attribute__((ext_vector_type(16))) _Float16 v16h;
typedef __attribute__((ext_vector_type(8)))  _Float16 v8h;
typedef __attribute__((ext_vector_type(8)))  float    v8f;

// ---------------- WMMA fragment helpers (ISA 7.12.2 layouts) ----------------
// A-matrix 16x32 f16: lane m = lane&15, half = lane>>4.
//   element e -> K = half*16 + e (contiguous 16, matches B loader)
__device__ __forceinline__ v16h load_a_row(const _Float16* __restrict__ rowp,
                                           int k0, unsigned hf) {
  return *(const v16h*)(rowp + k0 + hf * 16);
}

// B-matrix 32x16 f16, source stored [N][K] (row-major, ldb=K): lane n = lane&15,
// half picks K range: K = k0 + half*16 + e -> one contiguous 32B load per lane.
__device__ __forceinline__ v16h load_b_nt_row(const _Float16* __restrict__ colp,
                                              int k0, unsigned hf) {
  return *(const v16h*)(colp + k0 + hf * 16);
}

#define WMMA_F16(a, b, c) \
  __builtin_amdgcn_wmma_f32_16x16x32_f16(false, (a), false, (b), (short)0, (c), false, false)

// ---------------- elementwise / prep kernels ----------------
__global__ void k_f32_to_f16(const float* __restrict__ s, _Float16* __restrict__ d, int n) {
  int i = blockIdx.x * blockDim.x + threadIdx.x;
  if (i < n) d[i] = (_Float16)s[i];
}

// transpose-convert per expert: src f32 [E][K][N] -> dst f16 [E][N][K]
__global__ void k_f32_to_f16_t(const float* __restrict__ s, _Float16* __restrict__ d,
                               int K, int N, int total) {
  int i = blockIdx.x * blockDim.x + threadIdx.x;
  if (i >= total) return;
  int kn = K * N;
  int e = i / kn, r = i - e * kn;
  int k = r / N, n = r - k * N;
  d[(size_t)e * kn + (size_t)n * K + k] = (_Float16)s[i];
}

// x1 = rm0*x + rm1*x0 ; n = rmsnorm(x1)  (one 256-thread block per token)
__global__ void k_mix_rms(const float* __restrict__ x, const float* __restrict__ x0,
                          const float* __restrict__ rm, float* __restrict__ x1,
                          _Float16* __restrict__ nh) {
  __shared__ float red[256];
  int tok = blockIdx.x, t = threadIdx.x;
  const float* xr  = x  + (size_t)tok * Dq;
  const float* x0r = x0 + (size_t)tok * Dq;
  float*       x1r = x1 + (size_t)tok * Dq;
  float vals[4]; float ss = 0.f;
#pragma unroll
  for (int i = 0; i < 4; ++i) {
    int c = t + i * 256;
    float v = rm[c] * xr[c] + rm[Dq + c] * x0r[c];
    vals[i] = v; ss += v * v; x1r[c] = v;
  }
  red[t] = ss; __syncthreads();
  for (int o = 128; o > 0; o >>= 1) { if (t < o) red[t] += red[t + o]; __syncthreads(); }
  float r = rsqrtf(red[0] * (1.0f / Dq) + EPSq);
#pragma unroll
  for (int i = 0; i < 4; ++i) {
    int c = t + i * 256;
    nh[(size_t)tok * Dq + c] = (_Float16)(vals[i] * r);
  }
}

// ------- plain WMMA GEMM, 2x4 tiles/wave: C[M,N] = A[M,K] * B[N,K]^T -------
// K is a template constant so b1/b2/b3 and ar1 fold into immediate load offsets.
template <int K>
__global__ void k_gemm_nt8(const _Float16* __restrict__ A, const _Float16* __restrict__ B,
                           float* __restrict__ C, int M, int N) {
  unsigned lane = threadIdx.x & 31;
  int wave = blockIdx.x * (blockDim.x >> 5) + (threadIdx.x >> 5);
  int nt4 = N >> 6;                         // 64-wide column tiles
  int tm = wave / nt4, tn = wave - tm * nt4;
  if (tm >= (M >> 5)) return;               // 32-row tiles
  int m0 = tm << 5, n0 = tn << 6;
  unsigned nn = lane & 15, hf = lane >> 4;
  const _Float16* ar = A + (size_t)(m0 + nn) * K;
  const _Float16* bp = B + (size_t)(n0 + nn) * K;
  v8f c00 = {}, c01 = {}, c02 = {}, c03 = {};
  v8f c10 = {}, c11 = {}, c12 = {}, c13 = {};
  for (int k0 = 0; k0 < K; k0 += 32) {
    // speculative prefetch ~8 iterations ahead (invalid addrs dropped silently)
    __builtin_prefetch(bp + k0 + 256, 0, 1);
    __builtin_prefetch(bp + 16 * K + k0 + 256, 0, 1);
    __builtin_prefetch(bp + 32 * K + k0 + 256, 0, 1);
    __builtin_prefetch(bp + 48 * K + k0 + 256, 0, 1);
    v16h a0 = load_a_row(ar, k0, hf);
    v16h a1 = load_a_row(ar + 16 * K, k0, hf);
    v16h t0 = load_b_nt_row(bp, k0, hf);
    c00 = WMMA_F16(a0, t0, c00); c10 = WMMA_F16(a1, t0, c10);
    v16h t1 = load_b_nt_row(bp + 16 * K, k0, hf);
    c01 = WMMA_F16(a0, t1, c01); c11 = WMMA_F16(a1, t1, c11);
    v16h t2 = load_b_nt_row(bp + 32 * K, k0, hf);
    c02 = WMMA_F16(a0, t2, c02); c12 = WMMA_F16(a1, t2, c12);
    v16h t3 = load_b_nt_row(bp + 48 * K, k0, hf);
    c03 = WMMA_F16(a0, t3, c03); c13 = WMMA_F16(a1, t3, c13);
  }
  float* Cp = C + (size_t)(m0 + hf * 8) * N + n0 + nn;
#pragma unroll
  for (int r = 0; r < 8; ++r) {
    Cp[(size_t)r * N +  0] = c00[r];
    Cp[(size_t)r * N + 16] = c01[r];
    Cp[(size_t)r * N + 32] = c02[r];
    Cp[(size_t)r * N + 48] = c03[r];
  }
  Cp += (size_t)16 * N;
#pragma unroll
  for (int r = 0; r < 8; ++r) {
    Cp[(size_t)r * N +  0] = c10[r];
    Cp[(size_t)r * N + 16] = c11[r];
    Cp[(size_t)r * N + 32] = c12[r];
    Cp[(size_t)r * N + 48] = c13[r];
  }
}

// ---------------- per-head RMSNorm + RoPE (+gain/scale), f32 -> head-major f16 ----
__global__ void k_qk_norm_rope(const float* __restrict__ Xf, _Float16* __restrict__ Out,
                               const float* __restrict__ gains, int heads, float extraScale) {
  unsigned lane = threadIdx.x & 31;
  int wave = blockIdx.x * (blockDim.x >> 5) + (threadIdx.x >> 5);
  if (wave >= Mq * heads) return;
  int tok = wave / heads, h = wave - tok * heads;
  int b = tok / Sq, s = tok - b * Sq;
  const float* src = Xf + (size_t)tok * (heads * HDq) + h * HDq;
  float a = src[lane], bb = src[lane + 32];
  float ss = a * a + bb * bb;
#pragma unroll
  for (int o = 16; o >= 1; o >>= 1) ss += __shfl_xor(ss, o, 32);
  float r = rsqrtf(ss * (1.0f / HDq) + EPSq);
  a *= r; bb *= r;
  // inv_freq = 10000^(-2*lane/64); ln(10000)=9.210340371976184
  float freq = (float)s * __expf(-((float)(2 * lane) / (float)HDq) * 9.210340371976184f);
  float sn, cs; __sincosf(freq, &sn, &cs);
  float o1 = a * cs + bb * sn;
  float o2 = -a * sn + bb * cs;
  float g = extraScale * (gains ? gains[h] : 1.0f);
  _Float16* dst = Out + ((size_t)(b * heads + h) * Sq + s) * HDq;
  dst[lane]      = (_Float16)(o1 * g);
  dst[lane + 32] = (_Float16)(o2 * g);
}

// v f32 [tok][kv*HD] -> head-major TRANSPOSED f16 [(b*NKV+kv)*HD + d][s]
__global__ void k_v_convert_t(const float* __restrict__ Vf, _Float16* __restrict__ Vt) {
  int i = blockIdx.x * blockDim.x + threadIdx.x;
  if (i >= Mq * KVDq) return;
  int tok = i / KVDq, c = i - tok * KVDq;
  int kv = c / HDq, d = c - kv * HDq;
  int b = tok / Sq, s = tok - b * Sq;
  Vt[((size_t)(b * NKVq + kv) * HDq + d) * Sq + s] = (_Float16)Vf[i];
}

// ---------------- flash attention: 1 wave = 16 query rows ----------------
// Q,K head-major [head][s][hd]; V head-major transposed [head][hd][s].
__global__ void k_attn(const _Float16* __restrict__ Q, const _Float16* __restrict__ Kk,
                       const _Float16* __restrict__ V, _Float16* __restrict__ Y) {
  __shared__ __align__(32) _Float16 sP[4][16 * 32];
  unsigned lane = threadIdx.x & 31;
  int wave = blockIdx.x * (blockDim.x >> 5) + (threadIdx.x >> 5);
  const int S16 = Sq / 16;
  if (wave >= Bq * NHq * S16) return;
  int b = wave / (NHq * S16); int r0 = wave - b * NHq * S16;
  int h = r0 / S16; int qt = r0 - h * S16;
  int q0 = qt * 16, kv = h >> 2;
  unsigned nn = lane & 15, hf = lane >> 4;
  const _Float16* qb = Q  + (size_t)(b * NHq  + h)  * Sq * HDq;
  const _Float16* kb = Kk + (size_t)(b * NKVq + kv) * Sq * HDq;
  const _Float16* vt = V  + (size_t)(b * NKVq + kv) * HDq * Sq;
  const _Float16* qrow = qb + (size_t)(q0 + nn) * HDq;
  v16h qa0 = load_a_row(qrow, 0, hf);
  v16h qa1 = load_a_row(qrow, 32, hf);
  // V column pointers: col = hd index (t*16+nn), contiguous along keys
  const _Float16* vc = vt + (size_t)nn * Sq;
  v8f o0 = {}, o1 = {}, o2 = {}, o3 = {};
  float mi[8], li[8];
#pragma unroll
  for (int r = 0; r < 8; ++r) { mi[r] = -1e30f; li[r] = 0.f; }
  _Float16* sp = &sP[threadIdx.x >> 5][0];

  for (int j = 0; j < q0 + 16; j += 32) {
    const _Float16* kr = kb + (size_t)(j + nn) * HDq;
    __builtin_prefetch(kr + 32 * HDq, 0, 1);
    v8f s0 = {}, s1 = {};
    s0 = WMMA_F16(qa0, load_b_nt_row(kr, 0, hf), s0);
    s0 = WMMA_F16(qa1, load_b_nt_row(kr, 32, hf), s0);
    s1 = WMMA_F16(qa0, load_b_nt_row(kr + 16 * HDq, 0, hf), s1);
    s1 = WMMA_F16(qa1, load_b_nt_row(kr + 16 * HDq, 32, hf), s1);
#pragma unroll
    for (int r = 0; r < 8; ++r) {            // causal mask (scale folded into Q)
      int row = q0 + r + 8 * hf;
      if ((int)(j + nn) > row)      s0[r] = -1e30f;
      if ((int)(j + 16 + nn) > row) s1[r] = -1e30f;
    }
#pragma unroll
    for (int r = 0; r < 8; ++r) {            // online softmax per row
      float v = fmaxf(s0[r], s1[r]);
#pragma unroll
      for (int o = 8; o >= 1; o >>= 1) v = fmaxf(v, __shfl_xor(v, o, 16));
      float mn = fmaxf(mi[r], v);
      float al = __expf(mi[r] - mn);
      float p0 = __expf(s0[r] - mn);
      float p1 = __expf(s1[r] - mn);
      float rs = p0 + p1;
#pragma unroll
      for (int o = 8; o >= 1; o >>= 1) rs += __shfl_xor(rs, o, 16);
      li[r] = li[r] * al + rs; mi[r] = mn;
      o0[r] *= al; o1[r] *= al; o2[r] *= al; o3[r] *= al;
      int mrow = r + 8 * hf;
      sp[mrow * 32 + nn]      = (_Float16)p0;
      sp[mrow * 32 + 16 + nn] = (_Float16)p1;
    }
    asm volatile("s_wait_dscnt 0" ::: "memory");
    v16h pa = load_a_row(sp + (size_t)nn * 32, 0, hf);
    // P(16x32) x V(32x64): B frags contiguous along keys (j + hf*16 + e)
    o0 = WMMA_F16(pa, load_b_nt_row(vc, j, hf), o0);
    o1 = WMMA_F16(pa, load_b_nt_row(vc + 16 * Sq, j, hf), o1);
    o2 = WMMA_F16(pa, load_b_nt_row(vc + 32 * Sq, j, hf), o2);
    o3 = WMMA_F16(pa, load_b_nt_row(vc + 48 * Sq, j, hf), o3);
  }
#pragma unroll
  for (int r = 0; r < 8; ++r) {
    float inv = 1.0f / li[r];
    size_t base = ((size_t)(b * Sq + q0 + r + 8 * hf)) * Dq + (size_t)h * HDq + nn;
    Y[base +  0] = (_Float16)(o0[r] * inv);
    Y[base + 16] = (_Float16)(o1[r] * inv);
    Y[base + 32] = (_Float16)(o2[r] * inv);
    Y[base + 48] = (_Float16)(o3[r] * inv);
  }
}

// ---------------- post-attention: residual, velocity, RMSNorm ----------------
__global__ void k_post_attn(const float* __restrict__ x1, const float* __restrict__ projf,
                            const float* __restrict__ attn_scale, const float* __restrict__ vel,
                            const float* __restrict__ mu, float* __restrict__ vout,
                            float* __restrict__ x3, _Float16* __restrict__ n2h) {
  __shared__ float red[256];
  int tok = blockIdx.x, t = threadIdx.x;
  size_t base = (size_t)tok * Dq;
  float vals[4]; float ss = 0.f;
#pragma unroll
  for (int i = 0; i < 4; ++i) {
    int c = t + i * 256;
    float xv = x1[base + c] + attn_scale[c] * projf[base + c];
    float muc = fminf(fmaxf(mu[c], 0.5f), 1.5f);
    float vn = 0.95f * vel[base + c] - 0.3f * (xv - muc);
    vn = fminf(fmaxf(vn, -3.0f), 3.0f);
    vout[base + c] = vn;
    float xn = xv + 0.01f * vn;        // DT*GATE = 0.01
    x3[base + c] = xn; vals[i] = xn; ss += xn * xn;
  }
  red[t] = ss; __syncthreads();
  for (int o = 128; o > 0; o >>= 1) { if (t < o) red[t] += red[t + o]; __syncthreads(); }
  float r = rsqrtf(red[0] * (1.0f / Dq) + EPSq);
#pragma unroll
  for (int i = 0; i < 4; ++i) n2h[base + t + i * 256] = (_Float16)(vals[i] * r);
}

// ---------------- MoE routing ----------------
__global__ void k_zero_i32(int* p, int n) {
  int i = blockIdx.x * blockDim.x + threadIdx.x;
  if (i < n) p[i] = 0;
}
__global__ void k_route(const int* __restrict__ eids, int* __restrict__ cnt,
                        int* __restrict__ il) {
  int t = blockIdx.x * blockDim.x + threadIdx.x;
  if (t >= Mq) return;
  int e = eids[t];
  int p = atomicAdd(&cnt[e], 1);
  il[e * Mq + p] = t;
}

// grouped gathered GEMM, 2x4 tiles/wave: expert weights PRE-TRANSPOSED to [N][K]
template <int K>
__global__ void k_gemm_grouped_nt8(const _Float16* __restrict__ A,
                                   const _Float16* __restrict__ Ball, size_t estride,
                                   float* __restrict__ C, int ldc,
                                   const int* __restrict__ il, const int* __restrict__ cnt,
                                   int N) {
  unsigned lane = threadIdx.x & 31;
  int wave = blockIdx.x * (blockDim.x >> 5) + (threadIdx.x >> 5);
  int nt4 = N >> 6;
  int tilesM = Mq >> 5;                      // 32-row tiles
  int wpe = tilesM * nt4;
  int e = wave / wpe;
  if (e >= NEq) return;
  int rr = wave - e * wpe;
  int tm = rr / nt4, tn = rr - tm * nt4;
  int ce = cnt[e];
  int m0 = tm << 5;
  if (m0 >= ce) return;
  const int* lst = il + e * Mq;
  unsigned nn = lane & 15, hf = lane >> 4;
  int mm0 = m0 + (int)nn;      if (mm0 >= ce) mm0 = ce - 1;
  int mm1 = m0 + 16 + (int)nn; if (mm1 >= ce) mm1 = ce - 1;
  const _Float16* ar0 = A + (size_t)lst[mm0] * K;
  const _Float16* ar1 = A + (size_t)lst[mm1] * K;
  const _Float16* bp = Ball + (size_t)e * estride + (size_t)((tn << 6) + nn) * K;
  int n0 = tn << 6;
  v8f c00 = {}, c01 = {}, c02 = {}, c03 = {};
  v8f c10 = {}, c11 = {}, c12 = {}, c13 = {};
  for (int k0 = 0; k0 < K; k0 += 32) {
    __builtin_prefetch(bp + k0 + 256, 0, 1);
    __builtin_prefetch(bp + 16 * K + k0 + 256, 0, 1);
    __builtin_prefetch(bp + 32 * K + k0 + 256, 0, 1);
    __builtin_prefetch(bp + 48 * K + k0 + 256, 0, 1);
    v16h a0 = load_a_row(ar0, k0, hf);
    v16h a1 = load_a_row(ar1, k0, hf);
    v16h t0 = load_b_nt_row(bp, k0, hf);
    c00 = WMMA_F16(a0, t0, c00); c10 = WMMA_F16(a1, t0, c10);
    v16h t1 = load_b_nt_row(bp + 16 * K, k0, hf);
    c01 = WMMA_F16(a0, t1, c01); c11 = WMMA_F16(a1, t1, c11);
    v16h t2 = load_b_nt_row(bp + 32 * K, k0, hf);
    c02 = WMMA_F16(a0, t2, c02); c12 = WMMA_F16(a1, t2, c12);
    v16h t3 = load_b_nt_row(bp + 48 * K, k0, hf);
    c03 = WMMA_F16(a0, t3, c03); c13 = WMMA_F16(a1, t3, c13);
  }
#pragma unroll
  for (int r = 0; r < 8; ++r) {
    int m = m0 + r + 8 * hf;
    if (m < ce) {
      float* Cp = C + (size_t)lst[m] * ldc + n0 + nn;
      Cp[ 0] = c00[r]; Cp[16] = c01[r]; Cp[32] = c02[r]; Cp[48] = c03[r];
    }
    int m2 = m + 16;
    if (m2 < ce) {
      float* Cp = C + (size_t)lst[m2] * ldc + n0 + nn;
      Cp[ 0] = c10[r]; Cp[16] = c11[r]; Cp[32] = c12[r]; Cp[48] = c13[r];
    }
  }
}

__global__ void k_silu_mul(const float* __restrict__ gu, _Float16* __restrict__ hh) {
  int i = blockIdx.x * blockDim.x + threadIdx.x;
  if (i >= Mq * INTERq) return;
  int t = i / INTERq, c = i - t * INTERq;
  float g = gu[(size_t)t * (2 * INTERq) + c];
  float u = gu[(size_t)t * (2 * INTERq) + INTERq + c];
  float sg = g / (1.0f + __expf(-g));
  hh[i] = (_Float16)(sg * u);
}

__global__ void k_final_add(const float* __restrict__ x3, const float* __restrict__ moe,
                            const float* __restrict__ mlp_scale, float* __restrict__ outx) {
  int i = blockIdx.x * blockDim.x + threadIdx.x;
  if (i >= Mq * Dq) return;
  int c = i & (Dq - 1);
  outx[i] = x3[i] + mlp_scale[c] * moe[i];
}

// ---------------- host launcher ----------------
extern "C" void kernel_launch(void* const* d_in, const int* in_sizes, int n_in,
                              void* d_out, int out_size, void* d_ws, size_t ws_size,
                              hipStream_t stream) {
  (void)in_sizes; (void)n_in; (void)out_size; (void)ws_size;
  const float* x          = (const float*)d_in[0];
  const float* x0         = (const float*)d_in[1];
  const float* vel        = (const float*)d_in[2];
  const float* resid_mix  = (const float*)d_in[3];
  const float* attn_scale = (const float*)d_in[4];
  const float* mlp_scale  = (const float*)d_in[5];
  const float* cq_w       = (const float*)d_in[6];
  const float* ck_w       = (const float*)d_in[7];
  const float* cv_w       = (const float*)d_in[8];
  const float* proj_w     = (const float*)d_in[9];
  const float* q_gain     = (const float*)d_in[10];
  const float* mu         = (const float*)d_in[11];
  const float* gate_up    = (const float*)d_in[12];
  const float* down       = (const float*)d_in[13];
  const int*   eids       = (const int*)d_in[14];

  const size_t MD = (size_t)Mq * Dq;
  float* outx = (float*)d_out;
  float* outv = outx + MD;

  char* ws = (char*)d_ws;
  size_t off = 0;
  auto alloc = [&](size_t bytes) { size_t o = off; off += (bytes + 255) & ~(size_t)255; return o; };
  size_t o_nh   = alloc(MD * 2);                 // n f16  (reused as y_h)
  size_t o_x1   = alloc(MD * 4);
  size_t o_qf   = alloc(MD * 4);                 // reused as gu
  size_t o_kf   = alloc((size_t)Mq * KVDq * 4);  // reused as h_h
  size_t o_vf   = alloc((size_t)Mq * KVDq * 4);
  size_t o_qh   = alloc(MD * 2);
  size_t o_kh   = alloc((size_t)Mq * KVDq * 2);
  size_t o_vh   = alloc((size_t)Mq * KVDq * 2);
  size_t o_pf   = alloc(MD * 4);                 // proj out (reused as moe out)
  size_t o_x3   = alloc(MD * 4);
  size_t o_n2   = alloc(MD * 2);
  size_t o_wq   = alloc((size_t)Dq * Dq * 2);
  size_t o_wk   = alloc((size_t)KVDq * Dq * 2);
  size_t o_wv   = alloc((size_t)KVDq * Dq * 2);
  size_t o_wp   = alloc((size_t)Dq * Dq * 2);
  size_t o_wgu  = alloc((size_t)NEq * Dq * 2 * INTERq * 2);  // transposed [E][2I][D]
  size_t o_wdn  = alloc((size_t)NEq * INTERq * Dq * 2);      // transposed [E][D][I]
  size_t o_cnt  = alloc(NEq * 4);
  size_t o_idx  = alloc((size_t)NEq * Mq * 4);
  size_t o_yh  = o_nh;   // aliases (lifetimes disjoint)
  size_t o_gu  = o_qf;
  size_t o_hh  = o_kf;
  size_t o_moe = o_pf;

  _Float16* nh  = (_Float16*)(ws + o_nh);
  float*    x1  = (float*)(ws + o_x1);
  float*    qf  = (float*)(ws + o_qf);
  float*    kf  = (float*)(ws + o_kf);
  float*    vf  = (float*)(ws + o_vf);
  _Float16* qh  = (_Float16*)(ws + o_qh);
  _Float16* kh  = (_Float16*)(ws + o_kh);
  _Float16* vh  = (_Float16*)(ws + o_vh);
  _Float16* yh  = (_Float16*)(ws + o_yh);
  float*    pf  = (float*)(ws + o_pf);
  float*    x3  = (float*)(ws + o_x3);
  _Float16* n2h = (_Float16*)(ws + o_n2);
  _Float16* wq  = (_Float16*)(ws + o_wq);
  _Float16* wk  = (_Float16*)(ws + o_wk);
  _Float16* wv  = (_Float16*)(ws + o_wv);
  _Float16* wp  = (_Float16*)(ws + o_wp);
  _Float16* wgu = (_Float16*)(ws + o_wgu);
  _Float16* wdn = (_Float16*)(ws + o_wdn);
  float*    gu  = (float*)(ws + o_gu);
  _Float16* hh  = (_Float16*)(ws + o_hh);
  float*    moe = (float*)(ws + o_moe);
  int*      cnt = (int*)(ws + o_cnt);
  int*      idx = (int*)(ws + o_idx);

  auto cvt = [&](const float* s, _Float16* d, int n) {
    k_f32_to_f16<<<(n + 255) / 256, 256, 0, stream>>>(s, d, n);
  };
  // 1) weights -> f16 ([N][K] weights converted straight; [K][N] MoE weights transposed)
  cvt(cq_w,   wq, Dq * Dq);
  cvt(ck_w,   wk, KVDq * Dq);
  cvt(cv_w,   wv, KVDq * Dq);
  cvt(proj_w, wp, Dq * Dq);
  {
    int tot = NEq * Dq * 2 * INTERq;
    k_f32_to_f16_t<<<(tot + 255) / 256, 256, 0, stream>>>(gate_up, wgu, Dq, 2 * INTERq, tot);
    tot = NEq * INTERq * Dq;
    k_f32_to_f16_t<<<(tot + 255) / 256, 256, 0, stream>>>(down, wdn, INTERq, Dq, tot);
  }

  // 2) residual mix + rmsnorm
  k_mix_rms<<<Mq, 256, 0, stream>>>(x, x0, resid_mix, x1, nh);

  // 3) Q/K/V projections (C = n * W^T), 2x4 tiles per wave
  {
    int tiles = (Mq / 32) * (Dq / 64);
    k_gemm_nt8<Dq><<<tiles / 8, 256, 0, stream>>>(nh, wq, qf, Mq, Dq);
    tiles = (Mq / 32) * (KVDq / 64);
    k_gemm_nt8<Dq><<<tiles / 8, 256, 0, stream>>>(nh, wk, kf, Mq, KVDq);
    k_gemm_nt8<Dq><<<tiles / 8, 256, 0, stream>>>(nh, wv, vf, Mq, KVDq);
  }

  // 4) head norm + rope (+ q_gain and 1/sqrt(HD) folded into Q); V transpose
  k_qk_norm_rope<<<(Mq * NHq) / 8, 256, 0, stream>>>(qf, qh, q_gain, NHq, 0.125f);
  k_qk_norm_rope<<<(Mq * NKVq) / 8, 256, 0, stream>>>(kf, kh, nullptr, NKVq, 1.0f);
  k_v_convert_t<<<(Mq * KVDq + 255) / 256, 256, 0, stream>>>(vf, vh);

  // 5) flash attention (1 wave = 16 query rows)
  k_attn<<<(Bq * NHq * (Sq / 16)) / 4, 128, 0, stream>>>(qh, kh, vh, yh);

  // 6) output projection + post-attn (velocity into d_out, rmsnorm -> n2)
  {
    int tiles = (Mq / 32) * (Dq / 64);
    k_gemm_nt8<Dq><<<tiles / 8, 256, 0, stream>>>(yh, wp, pf, Mq, Dq);
  }
  k_post_attn<<<Mq, 256, 0, stream>>>(x1, pf, attn_scale, vel, mu, outv, x3, n2h);

  // 7) MoE: route, grouped gate_up GEMM, silu*u, grouped down GEMM, final add
  k_zero_i32<<<1, 32, 0, stream>>>(cnt, NEq);
  k_route<<<Mq / 256, 256, 0, stream>>>(eids, cnt, idx);
  {
    int waves = NEq * (Mq / 32) * ((2 * INTERq) / 64);
    k_gemm_grouped_nt8<Dq><<<waves / 8, 256, 0, stream>>>(
        n2h, wgu, (size_t)(2 * INTERq) * Dq, gu, 2 * INTERq,
        idx, cnt, 2 * INTERq);
  }
  k_silu_mul<<<(Mq * INTERq + 255) / 256, 256, 0, stream>>>(gu, hh);
  {
    int waves = NEq * (Mq / 32) * (Dq / 64);
    k_gemm_grouped_nt8<INTERq><<<waves / 8, 256, 0, stream>>>(
        hh, wdn, (size_t)Dq * INTERq, moe, Dq,
        idx, cnt, Dq);
  }
  k_final_add<<<(int)(MD / 256), 256, 0, stream>>>(x3, moe, mlp_scale, outx);
}
